// MambaSequenceClassifier_2534030705277
// MI455X (gfx1250) — compile-verified, matrix operand order checked by reading, not run on hardware
//
#include <hip/hip_runtime.h>
#include <cstddef>
#include <cstdint>

// ---- model dims (match reference) ----
#define D_MODEL 256
#define N_LAYERS 2
#define D_INNER 512
#define D_STATE 16
#define D_CONV 4
#define DT_RANK 16
#define BATCH 8
#define SEQ 1024
#define EPSV 1e-5f
#define TOKENS (BATCH * SEQ)          // 8192
#define DBC_W (DT_RANK + 2 * D_STATE) // 48

typedef __attribute__((ext_vector_type(16))) _Float16 v16h;
typedef __attribute__((ext_vector_type(8)))  float    v8f;
typedef unsigned int __attribute__((ext_vector_type(4))) u32x4;
typedef int          __attribute__((ext_vector_type(4))) i32x4;
typedef int          __attribute__((ext_vector_type(8))) i32x8;

__device__ __forceinline__ float sigmoidf_(float x) { return 1.0f / (1.0f + __expf(-x)); }
__device__ __forceinline__ float siluf_(float x)    { return x * sigmoidf_(x); }
__device__ __forceinline__ float softplusf_(float x){ return (x > 20.0f) ? x : log1pf(__expf(x)); }

// ---------------------------------------------------------------------------
// Weight pre-swizzle: f32 B[K,N] -> f16 fragments, layout [kt][nt][lane][16h]
// so each wave's B operand is one contiguous 32-byte v16h per lane.
// Zero-pads K up to Kt*32 (bakes in the K=16 delta-projection padding).
// ---------------------------------------------------------------------------
__global__ void swizzle_b_kernel(const float* __restrict__ B, _Float16* __restrict__ out,
                                 int K, int N, int Kt) {
  int tilesN = N >> 4;
  int total = Kt * tilesN * 32;
  int idx = blockIdx.x * blockDim.x + threadIdx.x;
  if (idx >= total) return;
  int lane = idx & 31;
  int tile = idx >> 5;
  int kt = tile / tilesN;
  int nt = tile % tilesN;
  int c  = nt * 16 + (lane & 15);
  int kb = kt * 32 + ((lane >> 4) << 3);
  _Float16* o = out + (size_t)idx * 16;
#pragma unroll
  for (int i = 0; i < 8; ++i) {
    int ka = kb + i;
    int kc = kb + 16 + i;
    o[i]     = (ka < K) ? (_Float16)B[(size_t)ka * N + c] : (_Float16)0.0f;
    o[i + 8] = (kc < K) ? (_Float16)B[(size_t)kc * N + c] : (_Float16)0.0f;
  }
}

// ---------------------------------------------------------------------------
// A fragment: no bounds checks, 4x float4 vector loads (b128).
// ---------------------------------------------------------------------------
__device__ __forceinline__ v16h load_a_frag_fast(const float* __restrict__ A, int row0, int k0,
                                                 int lda, int lane) {
  int r  = row0 + (lane & 15);
  int kb = k0 + ((lane >> 4) << 3);
  const float* p = A + (size_t)r * lda + kb;
  float4 f0 = ((const float4*)p)[0];
  float4 f1 = ((const float4*)p)[1];
  float4 f2 = ((const float4*)(p + 16))[0];
  float4 f3 = ((const float4*)(p + 16))[1];
  v16h out;
  out[0]  = (_Float16)f0.x; out[1]  = (_Float16)f0.y; out[2]  = (_Float16)f0.z; out[3]  = (_Float16)f0.w;
  out[4]  = (_Float16)f1.x; out[5]  = (_Float16)f1.y; out[6]  = (_Float16)f1.z; out[7]  = (_Float16)f1.w;
  out[8]  = (_Float16)f2.x; out[9]  = (_Float16)f2.y; out[10] = (_Float16)f2.z; out[11] = (_Float16)f2.w;
  out[12] = (_Float16)f3.x; out[13] = (_Float16)f3.y; out[14] = (_Float16)f3.z; out[15] = (_Float16)f3.w;
  return out;
}

// ---------------------------------------------------------------------------
// WMMA GEMM with LDS-staged B via the Tensor Data Mover.
// Block = 4 waves = 4 M-tiles sharing one N-group of NSUB tiles.
// The block's whole B strip ([Kt][NSUB*512] halves, row stride tilesN*512)
// is DMA'd into LDS once, then each wave runs Kt x NSUB WMMAs.  All NSUB
// B fragments are preloaded into distinct registers per k-step so the
// ds_load latency overlaps and the WMMAs issue back-to-back.
// EP=0: store   EP=1: softplus(c + bias[col])   EP=2: c + res[row,col]
// ---------------------------------------------------------------------------
template <int EP, int NSUB>
__global__ void wmma_gemm_lds_kernel(const float* __restrict__ A, const _Float16* __restrict__ Bs,
                                     const float* __restrict__ bias, const float* __restrict__ res,
                                     float* __restrict__ C,
                                     int M, int N, int Kt, int lda, int ldc) {
  extern __shared__ _Float16 bsh[];
  int lane = threadIdx.x & 31;
  int wave = threadIdx.x >> 5;
  int tilesN  = N >> 4;
  int ngroups = tilesN / NSUB;
  int ngrp = blockIdx.x % ngroups;
  int rb   = blockIdx.x / ngroups;
  int row0 = (rb * 4 + wave) << 4;
  int ntg  = ngrp * NSUB;

  const int rowHalves = NSUB * 512;               // halves per kt row of the strip
  const _Float16* gsrc = Bs + (size_t)ntg * 512;  // strip row 0 (kt=0)

#if defined(__HIP_DEVICE_COMPILE__) && __has_builtin(__builtin_amdgcn_tensor_load_to_lds)
  if (wave == 0) {
    unsigned ldsAddr = (unsigned)(uintptr_t)&bsh[0];
    unsigned long long ga = (unsigned long long)(uintptr_t)gsrc;
    // ---- D# group 0: count=1 | lds_addr | global_addr | type=2 ----
    u32x4 g0;
    g0[0] = 1u;                                     // count=1, user mode
    g0[1] = ldsAddr;                                // lds_addr (bytes)
    g0[2] = (unsigned)(ga & 0xffffffffu);           // global_addr[31:0]
    g0[3] = (unsigned)((ga >> 32) & 0x1ffffffu) | (2u << 30); // addr[56:32] | type=2
    // ---- D# group 1: 2-D tile, data_size=2B ----
    unsigned td0 = (unsigned)rowHalves;             // tensor_dim0
    unsigned td1 = (unsigned)Kt;                    // tensor_dim1
    long long st0 = (long long)tilesN * 512;        // tensor_dim0_stride (elems)
    i32x8 g1;
    g1[0] = (int)(1u << 16);                        // data_size=1 -> 2 bytes
    g1[1] = (int)((td0 & 0xffffu) << 16);           // tensor_dim0[15:0]
    g1[2] = (int)(((td0 >> 16) & 0xffffu) | ((td1 & 0xffffu) << 16));
    g1[3] = (int)(((td1 >> 16) & 0xffffu) | (((unsigned)rowHalves & 0xffffu) << 16)); // tile_dim0
    g1[4] = (int)((unsigned)Kt & 0xffffu);          // tile_dim1 (tile_dim2=0)
    g1[5] = (int)(st0 & 0xffffffffll);              // stride0[31:0]
    g1[6] = (int)((st0 >> 32) & 0xffffll);          // stride0[47:32]
    g1[7] = 0;
    i32x4 z4 = {};
#if __has_include(<hip/amd_detail/amd_gfx1250_TDM.h>)
    i32x8 z8 = {};
    __builtin_amdgcn_tensor_load_to_lds(g0, g1, z4, z4, z8, 0);
#else
    __builtin_amdgcn_tensor_load_to_lds(g0, g1, z4, z4, 0);
#endif
    __builtin_amdgcn_s_wait_tensorcnt(0);
  }
  __syncthreads();
#else
  // Cooperative fallback: copy strip in 16B chunks.
  {
    int chunks = (Kt * rowHalves) >> 3;     // uint4 = 8 halves
    int perRow = rowHalves >> 3;
    for (int i = threadIdx.x; i < chunks; i += blockDim.x) {
      int ktRow = i / perRow;
      int off   = i % perRow;
      ((uint4*)bsh)[i] =
          ((const uint4*)(gsrc + (size_t)ktRow * tilesN * 512))[off];
    }
  }
  __syncthreads();
#endif

  v8f acc[NSUB];
#pragma unroll
  for (int n = 0; n < NSUB; ++n) acc[n] = (v8f){};

  for (int kt = 0; kt < Kt; ++kt) {
    v16h a = load_a_frag_fast(A, row0, kt * 32, lda, lane);
    // preload all NSUB B fragments into distinct registers (overlap DS latency)
    v16h bfr[NSUB];
#pragma unroll
    for (int n = 0; n < NSUB; ++n)
      bfr[n] = *(const v16h*)(bsh + ((size_t)(kt * NSUB + n) * 32 + lane) * 16);
#pragma unroll
    for (int n = 0; n < NSUB; ++n)
      acc[n] = __builtin_amdgcn_wmma_f32_16x16x32_f16(false, a, false, bfr[n], (short)0, acc[n],
                                                      false, false);
  }

  int rbase = row0 + ((lane >> 4) << 3);
  int colb  = lane & 15;
#pragma unroll
  for (int n = 0; n < NSUB; ++n) {
    int col = (ntg + n) * 16 + colb;
#pragma unroll
    for (int r = 0; r < 8; ++r) {
      int row = rbase + r;
      float v = acc[n][r];
      if (EP == 1) v = softplusf_(v + bias[col]);
      else if (EP == 2) v += res[(size_t)row * ldc + col];
      C[(size_t)row * ldc + col] = v;
    }
  }
}

// ---------------------------------------------------------------------------
// RMSNorm over rows of width D_MODEL (blockDim = 256 = D_MODEL)
// ---------------------------------------------------------------------------
__global__ void rmsnorm_kernel(const float* __restrict__ x, const float* __restrict__ w,
                               float* __restrict__ out, int d) {
  __shared__ float red[D_MODEL];
  int row = blockIdx.x;
  int tid = threadIdx.x;
  float xv = x[(size_t)row * d + tid];
  red[tid] = xv * xv;
  __syncthreads();
  for (int s = d >> 1; s > 0; s >>= 1) {
    if (tid < s) red[tid] += red[tid + s];
    __syncthreads();
  }
  float scale = rsqrtf(red[0] / (float)d + EPSV);
  out[(size_t)row * d + tid] = xv * scale * w[tid];
}

// ---------------------------------------------------------------------------
// Causal depthwise conv (width 4) + bias + SiLU.
// ---------------------------------------------------------------------------
__global__ void conv_silu_kernel(const float* __restrict__ xz, const float* __restrict__ cw,
                                 const float* __restrict__ cb, float* __restrict__ xi) {
  int idx = blockIdx.x * blockDim.x + threadIdx.x;
  if (idx >= TOKENS * D_INNER) return;
  int e   = idx % D_INNER;
  int tok = idx / D_INNER;
  int t = tok % SEQ;
  int b = tok / SEQ;
  float acc = cb[e];
#pragma unroll
  for (int j = 0; j < D_CONV; ++j) {
    int tt = t - (D_CONV - 1) + j;
    if (tt >= 0) acc += cw[e * D_CONV + j] * xz[((size_t)(b * SEQ + tt)) * (2 * D_INNER) + e];
  }
  xi[(size_t)tok * D_INNER + e] = siluf_(acc);
}

// ---------------------------------------------------------------------------
// Selective scan: one lane per (batch, channel, state) -> 65536 lanes.
// 16-lane shfl_xor reduction for the h . C dot each step.
// ---------------------------------------------------------------------------
__global__ void scan_kernel(const float* __restrict__ xi, const float* __restrict__ delta,
                            const float* __restrict__ dbc, const float* __restrict__ alog,
                            const float* __restrict__ dsk, float* __restrict__ y) {
  int idx = blockIdx.x * blockDim.x + threadIdx.x;
  if (idx >= BATCH * D_INNER * D_STATE) return;
  int n  = idx & (D_STATE - 1);
  int ce = idx >> 4;
  int e  = ce % D_INNER;
  int b  = ce / D_INNER;
  float An = -__expf(alog[e * D_STATE + n]);
  float Dv = dsk[e];
  float h = 0.0f;
  for (int t = 0; t < SEQ; ++t) {
    size_t roff = (size_t)(b * SEQ + t);
    float dlt = delta[roff * D_INNER + e];
    float u   = xi[roff * D_INNER + e];
    const float* row = dbc + roff * DBC_W;
    h = __expf(dlt * An) * h + (dlt * u) * row[DT_RANK + n];
    float yv = h * row[DT_RANK + D_STATE + n];
    yv += __shfl_xor(yv, 1, 16);
    yv += __shfl_xor(yv, 2, 16);
    yv += __shfl_xor(yv, 4, 16);
    yv += __shfl_xor(yv, 8, 16);
    if (n == 0) y[roff * D_INNER + e] = yv + u * Dv;
  }
}

// ---------------------------------------------------------------------------
// Gating: yg = y * silu(z)
// ---------------------------------------------------------------------------
__global__ void gate_kernel(const float* __restrict__ y, const float* __restrict__ xz,
                            float* __restrict__ yg) {
  int idx = blockIdx.x * blockDim.x + threadIdx.x;
  if (idx >= TOKENS * D_INNER) return;
  int e   = idx % D_INNER;
  int tok = idx / D_INNER;
  float z = xz[(size_t)tok * (2 * D_INNER) + D_INNER + e];
  yg[idx] = y[idx] * siluf_(z);
}

// ---------------------------------------------------------------------------
// Final: rmsnorm of last token per batch, dot with fc_w, + fc_b
// ---------------------------------------------------------------------------
__global__ void final_kernel(const float* __restrict__ h, const float* __restrict__ nfw,
                             const float* __restrict__ fcw, const float* __restrict__ fcb,
                             float* __restrict__ out) {
  __shared__ float red[D_MODEL];
  int b = blockIdx.x;
  int tid = threadIdx.x;
  float hv = h[((size_t)b * SEQ + (SEQ - 1)) * D_MODEL + tid];
  red[tid] = hv * hv;
  __syncthreads();
  for (int s = D_MODEL >> 1; s > 0; s >>= 1) {
    if (tid < s) red[tid] += red[tid + s];
    __syncthreads();
  }
  float scale = rsqrtf(red[0] / (float)D_MODEL + EPSV);
  float val = hv * scale * nfw[tid] * fcw[tid];
  __syncthreads();
  red[tid] = val;
  __syncthreads();
  for (int s = D_MODEL >> 1; s > 0; s >>= 1) {
    if (tid < s) red[tid] += red[tid + s];
    __syncthreads();
  }
  if (tid == 0) out[b] = red[0] + fcb[0];
}

// ---------------------------------------------------------------------------
// Host launcher
// ---------------------------------------------------------------------------
static inline int ceil_div(int a, int b) { return (a + b - 1) / b; }

extern "C" void kernel_launch(void* const* d_in, const int* in_sizes, int n_in,
                              void* d_out, int out_size, void* d_ws, size_t ws_size,
                              hipStream_t stream) {
  (void)in_sizes; (void)n_in; (void)out_size; (void)ws_size;

  const float* x      = (const float*)d_in[0];
  const float* norm_w = (const float*)d_in[1];
  const float* ipw    = (const float*)d_in[2];
  const float* cw     = (const float*)d_in[3];
  const float* cb     = (const float*)d_in[4];
  const float* xpw    = (const float*)d_in[5];
  const float* dtw    = (const float*)d_in[6];
  const float* dtb    = (const float*)d_in[7];
  const float* alog   = (const float*)d_in[8];
  const float* dsk    = (const float*)d_in[9];
  const float* opw    = (const float*)d_in[10];
  const float* nfw    = (const float*)d_in[11];
  const float* fcw    = (const float*)d_in[12];
  const float* fcb    = (const float*)d_in[13];
  float* out = (float*)d_out;

  // ---- workspace layout ----
  float* ws    = (float*)d_ws;
  float* xn    = ws;
  float* xz    = xn    + (size_t)TOKENS * D_MODEL;
  float* xi    = xz    + (size_t)TOKENS * 2 * D_INNER;
  float* dbc   = xi    + (size_t)TOKENS * D_INNER;
  float* delta = dbc   + (size_t)TOKENS * DBC_W;
  float* yb    = delta + (size_t)TOKENS * D_INNER;
  float* yg    = yb    + (size_t)TOKENS * D_INNER;
  float* h0    = yg    + (size_t)TOKENS * D_INNER;
  float* h1    = h0    + (size_t)TOKENS * D_MODEL;

  const int KT_IP = D_MODEL / 32;          // 8
  const int KT_XP = D_INNER / 32;          // 16
  const int KT_DT = 1;                     // K=16 zero-padded to 32
  const int KT_OP = D_INNER / 32;          // 16
  const size_t SZ_IP = (size_t)KT_IP * (2 * D_INNER / 16) * 512;
  const size_t SZ_XP = (size_t)KT_XP * (DBC_W / 16)       * 512;
  const size_t SZ_DT = (size_t)KT_DT * (D_INNER / 16)     * 512;
  const size_t SZ_OP = (size_t)KT_OP * (D_MODEL / 16)     * 512;
  _Float16* f16p = (_Float16*)(h1 + (size_t)TOKENS * D_MODEL);
  _Float16 *ipw_s[N_LAYERS], *xpw_s[N_LAYERS], *dtw_s[N_LAYERS], *opw_s[N_LAYERS];
  for (int l = 0; l < N_LAYERS; ++l) {
    ipw_s[l] = f16p; f16p += SZ_IP;
    xpw_s[l] = f16p; f16p += SZ_XP;
    dtw_s[l] = f16p; f16p += SZ_DT;
    opw_s[l] = f16p; f16p += SZ_OP;
  }

  const int SWZ_BLOCK = 256;
  for (int l = 0; l < N_LAYERS; ++l) {
    int n;
    n = KT_IP * (2 * D_INNER / 16) * 32;
    swizzle_b_kernel<<<ceil_div(n, SWZ_BLOCK), SWZ_BLOCK, 0, stream>>>(
        ipw + (size_t)l * D_MODEL * 2 * D_INNER, ipw_s[l], D_MODEL, 2 * D_INNER, KT_IP);
    n = KT_XP * (DBC_W / 16) * 32;
    swizzle_b_kernel<<<ceil_div(n, SWZ_BLOCK), SWZ_BLOCK, 0, stream>>>(
        xpw + (size_t)l * D_INNER * DBC_W, xpw_s[l], D_INNER, DBC_W, KT_XP);
    n = KT_DT * (D_INNER / 16) * 32;
    swizzle_b_kernel<<<ceil_div(n, SWZ_BLOCK), SWZ_BLOCK, 0, stream>>>(
        dtw + (size_t)l * DT_RANK * D_INNER, dtw_s[l], DT_RANK, D_INNER, KT_DT);
    n = KT_OP * (D_MODEL / 16) * 32;
    swizzle_b_kernel<<<ceil_div(n, SWZ_BLOCK), SWZ_BLOCK, 0, stream>>>(
        opw + (size_t)l * D_INNER * D_MODEL, opw_s[l], D_INNER, D_MODEL, KT_OP);
  }

  const int GEMM_BLOCK = 128;   // 4 waves = 4 M-tiles per block
  const int EW_BLOCK = 256;
  int nEW = TOKENS * D_INNER;
  int ewBlocks = ceil_div(nEW, EW_BLOCK);
  const int ROWB = TOKENS / 64; // 128 row-blocks of 4 M-tiles

  const float* cur = x;
  float* hbuf[2] = {h0, h1};

  for (int l = 0; l < N_LAYERS; ++l) {
    rmsnorm_kernel<<<TOKENS, D_MODEL, 0, stream>>>(cur, norm_w + l * D_MODEL, xn, D_MODEL);

    // xz = xn @ in_proj_w[l]   (N=1024, K=256, NSUB=4; LDS = 8*4KB = 32KB)
    {
      int ngroups = (2 * D_INNER / 16) / 4;
      wmma_gemm_lds_kernel<0, 4><<<ROWB * ngroups, GEMM_BLOCK, KT_IP * 4 * 1024, stream>>>(
          xn, ipw_s[l], nullptr, nullptr, xz, TOKENS, 2 * D_INNER, KT_IP, D_MODEL, 2 * D_INNER);
    }

    conv_silu_kernel<<<ewBlocks, EW_BLOCK, 0, stream>>>(
        xz, cw + (size_t)l * D_INNER * D_CONV, cb + (size_t)l * D_INNER, xi);

    // dbc = xi @ x_proj_w[l]   (N=48, K=512, NSUB=3; LDS = 16*3KB = 48KB)
    {
      int ngroups = (DBC_W / 16) / 3;
      wmma_gemm_lds_kernel<0, 3><<<ROWB * ngroups, GEMM_BLOCK, KT_XP * 3 * 1024, stream>>>(
          xi, xpw_s[l], nullptr, nullptr, dbc, TOKENS, DBC_W, KT_XP, D_INNER, DBC_W);
    }

    // delta = softplus(dt_r @ dt_proj_w[l] + dtb)  (K=16 zero-padded; LDS = 4KB)
    {
      int ngroups = (D_INNER / 16) / 4;
      wmma_gemm_lds_kernel<1, 4><<<ROWB * ngroups, GEMM_BLOCK, KT_DT * 4 * 1024, stream>>>(
          dbc, dtw_s[l], dtb + (size_t)l * D_INNER, nullptr, delta,
          TOKENS, D_INNER, KT_DT, DBC_W, D_INNER);
    }

    {
      int nThreads = BATCH * D_INNER * D_STATE;
      scan_kernel<<<ceil_div(nThreads, EW_BLOCK), EW_BLOCK, 0, stream>>>(
          xi, delta, dbc, alog + (size_t)l * D_INNER * D_STATE, dsk + (size_t)l * D_INNER, yb);
    }

    gate_kernel<<<ewBlocks, EW_BLOCK, 0, stream>>>(yb, xz, yg);

    // h = cur + yg @ out_proj_w[l]  (N=256, K=512, NSUB=4; LDS = 16*4KB = 64KB)
    {
      int ngroups = (D_MODEL / 16) / 4;
      wmma_gemm_lds_kernel<2, 4><<<ROWB * ngroups, GEMM_BLOCK, KT_OP * 4 * 1024, stream>>>(
          yg, opw_s[l], nullptr, cur, hbuf[l], TOKENS, D_MODEL, KT_OP, D_INNER, D_MODEL);
    }

    cur = hbuf[l];
  }

  final_kernel<<<BATCH, D_MODEL, 0, stream>>>(cur, nfw, fcw, fcb, out);
}
